// BahdanauAttention_16518444220431
// MI455X (gfx1250) — matrix operand order repaired and checked by reading, not compiled, not run on hardware
//
#include <hip/hip_runtime.h>
#include <hip/hip_bf16.h>

typedef __bf16 bf16_t;
typedef __attribute__((ext_vector_type(16))) __bf16 v16bf;
typedef __attribute__((ext_vector_type(8)))  __bf16 v8bf;
typedef __attribute__((ext_vector_type(8)))  float  v8f;

#define SEQ_L   2048
#define DIM     256
#define BATCH   16
#define NROWS   (BATCH * SEQ_L)   /* 32768 rows (b,l) */
#define T_ITERS 8

// ---------------------------------------------------------------------------
// Pack W (256x256 fp32, W[o][k]) into bf16 WMMA B-fragment layout for
// v_wmma_f32_16x16x32_bf16:  B[k][n] = W[n][k].
// Fragment layout (CDNA5 ISA, 16-bit B 32x16): lane t holds column
// n = t&15; half h -> k = (t>>4)*8 + (h&7) + (h>>3)*16. One fragment = 32 lanes * 16 halves,
// stored contiguously so each lane loads its 32 bytes with one vector load.
// frag index = ((ko*16 + no)*32 + lane)*16 + h
// ---------------------------------------------------------------------------
__global__ void pack_wfrag_kernel(const float* __restrict__ W,
                                  bf16_t* __restrict__ frag) {
  const int j    = blockIdx.x * 256 + threadIdx.x;   // 0 .. 65535
  const int h    = j & 15;
  const int t    = (j >> 4) & 31;
  const int tile = j >> 9;                           // ko*16 + no
  const int ko   = tile >> 4;
  const int no   = tile & 15;
  const int n    = no * 16 + (t & 15);
  const int k    = ko * 32 + ((t >> 4) * 8) + (h & 7) + ((h >> 3) * 16);
  frag[j] = (bf16_t)W[n * DIM + k];
}

__global__ void zero_u4_kernel(uint4* __restrict__ p, int n) {
  const int j = blockIdx.x * 256 + threadIdx.x;
  if (j < n) p[j] = make_uint4(0u, 0u, 0u, 0u);
}

// Unconditional 16x32 bf16 A-fragment load (ISA layout: lane row m = t&15,
// halves 0..7 -> K = (t>>4)*8 + h, halves 8..15 -> K = (t>>4)*8 + 16 + h).
// Out-of-range shift rows are handled by pointing `base` at a zeroed dummy
// row (per-lane v_cndmask on the address) so there is no EXEC divergence and
// no per-K-step zero rebuild.
static __device__ __forceinline__ v16bf load_a_frag(const bf16_t* base) {
  union { v16bf v; v8bf h[2]; } u;
  u.h[0] = *(const v8bf*)(base);
  u.h[1] = *(const v8bf*)(base + 16);
  return u.v;
}

// ---------------------------------------------------------------------------
// One DP iteration:  next[dir] = relu(a + shift(prev[dir]) @ W2^T + b2)
// dir 0 = "up"   (reads row r-1, zero at l==0)
// dir 1 = "down" (reads row r+1, zero at l==L-1)
// Block: 64 rows (4 M-subtiles), 256 threads = 8 waves.
// Waves 0-3 -> up, 4-7 -> down; each wave owns 4 N-tiles of 16 cols.
// B fragments are held in registers across the 4 M-subtiles -> each wave
// issues 128 WMMAs per block with ~32 flops/byte of cache traffic.
// The  a + b1 + b2  term is recomputed inline (dims=4).
// ---------------------------------------------------------------------------
__global__ __launch_bounds__(256) void dp_iter_kernel(
    const bf16_t* __restrict__ prev,   // [2][NROWS][DIM] bf16
    bf16_t* __restrict__ next,         // [2][NROWS][DIM] bf16
    const bf16_t* __restrict__ wfrag,  // W2 fragments
    const bf16_t* __restrict__ zrow,   // 256 zeroed bf16 (shift boundary)
    const float* __restrict__ x,       // [BATCH][4][SEQ_L]
    const float* __restrict__ W1,      // [DIM][4]
    const float* __restrict__ b1,
    const float* __restrict__ b2) {
  const int t    = threadIdx.x & 31;
  const int w    = threadIdx.x >> 5;
  const int dir  = w >> 2;
  const int q    = w & 3;
  const int hi   = t >> 4;
  const int mrow = t & 15;
  const int row0 = blockIdx.x * 64;

  const bf16_t* pd = prev + (size_t)dir * NROWS * DIM;
  const bf16_t* abase[4];
#pragma unroll
  for (int m = 0; m < 4; ++m) {
    const int r = row0 + m * 16 + mrow;
    const int l = r & (SEQ_L - 1);
    const bool valid = dir ? (l != SEQ_L - 1) : (l != 0);
    const long srcR  = dir ? (long)r + 1 : (long)r - 1;
    abase[m] = valid ? (pd + srcR * DIM + hi * 8) : (zrow + hi * 8);
  }

  v8f acc[4][4] = {};                      // [mtile][ntile]
#pragma unroll
  for (int ko = 0; ko < 8; ++ko) {
    v16bf bfrag[4];
#pragma unroll
    for (int j = 0; j < 4; ++j)
      bfrag[j] =
          *(const v16bf*)(wfrag + (size_t)((ko * 16 + (q * 4 + j)) * 32 + t) * 16);
#pragma unroll
    for (int m = 0; m < 4; ++m) {
      const v16bf afrag = load_a_frag(abase[m] + ko * 32);
#pragma unroll
      for (int j = 0; j < 4; ++j)
        acc[m][j] = __builtin_amdgcn_wmma_f32_16x16x32_bf16(
            false, afrag, false, bfrag[j], (short)0, acc[m][j], false, false);
    }
  }

  // Epilogue: add recomputed  a = x.W1^T + b1 (+b2), relu, store bf16.
  bf16_t* nbase = next + (size_t)dir * NROWS * DIM;
  float w1v[4][4], bias[4];
  int ncol[4];
#pragma unroll
  for (int j = 0; j < 4; ++j) {
    const int n = (q * 4 + j) * 16 + mrow;           // C/D col = t&15
    ncol[j] = n;
#pragma unroll
    for (int dd = 0; dd < 4; ++dd) w1v[j][dd] = W1[n * 4 + dd];
    bias[j] = b1[n] + b2[n];
  }
#pragma unroll
  for (int m = 0; m < 4; ++m) {
#pragma unroll
    for (int i = 0; i < 8; ++i) {                    // C/D row = i + 8*(t>>4)
      const int rr = row0 + m * 16 + hi * 8 + i;
      const int ll = rr & (SEQ_L - 1);
      const int bb = rr >> 11;                       // / SEQ_L
      const float* xp = x + (size_t)(bb * 4) * SEQ_L + ll;
      const float x0 = xp[0], x1 = xp[SEQ_L], x2 = xp[2 * SEQ_L], x3 = xp[3 * SEQ_L];
#pragma unroll
      for (int j = 0; j < 4; ++j) {
        float v = acc[m][j][i] + bias[j] + x0 * w1v[j][0] + x1 * w1v[j][1] +
                  x2 * w1v[j][2] + x3 * w1v[j][3];
        v = v > 0.f ? v : 0.f;
        nbase[(size_t)rr * DIM + ncol[j]] = (bf16_t)v;
      }
    }
  }
}

// ---------------------------------------------------------------------------
// Final:  miu = relu(c + shift_r(up)@W4^T + shift_l(down)@W4^T + 2*b4)
// Block: 32 rows (2 M-subtiles). Each wave owns 2 N-tiles; each B fragment
// feeds 4 WMMAs (2 M-subtiles x both shift directions) -> 64 WMMAs per wave.
// ---------------------------------------------------------------------------
__global__ __launch_bounds__(256) void dp_final_kernel(
    const bf16_t* __restrict__ buf,    // [2][NROWS][DIM] (up, down) after T iters
    const bf16_t* __restrict__ wfrag,  // W4 fragments
    const bf16_t* __restrict__ zrow,
    const float* __restrict__ x,
    const float* __restrict__ W3,      // [DIM][4]
    const float* __restrict__ b3,
    const float* __restrict__ b4,
    float* __restrict__ out) {
  const int t    = threadIdx.x & 31;
  const int w    = threadIdx.x >> 5;
  const int hi   = t >> 4;
  const int mrow = t & 15;
  const int row0 = blockIdx.x * 32;

  const bf16_t* ub[2];
  const bf16_t* db[2];
#pragma unroll
  for (int m = 0; m < 2; ++m) {
    const int r = row0 + m * 16 + mrow;
    const int l = r & (SEQ_L - 1);
    ub[m] = (l != 0)         ? (buf + (long)(r - 1) * DIM + hi * 8)
                             : (zrow + hi * 8);
    db[m] = (l != SEQ_L - 1) ? (buf + (long)NROWS * DIM + (long)(r + 1) * DIM + hi * 8)
                             : (zrow + hi * 8);
  }

  v8f acc[2][2] = {};                      // [mtile][ntile]
#pragma unroll
  for (int ko = 0; ko < 8; ++ko) {
    v16bf bfrag[2];
#pragma unroll
    for (int j = 0; j < 2; ++j)
      bfrag[j] =
          *(const v16bf*)(wfrag + (size_t)((ko * 16 + (w * 2 + j)) * 32 + t) * 16);
#pragma unroll
    for (int m = 0; m < 2; ++m) {
      const v16bf au = load_a_frag(ub[m] + ko * 32);
      const v16bf ad = load_a_frag(db[m] + ko * 32);
#pragma unroll
      for (int j = 0; j < 2; ++j) {
        acc[m][j] = __builtin_amdgcn_wmma_f32_16x16x32_bf16(
            false, au, false, bfrag[j], (short)0, acc[m][j], false, false);
        acc[m][j] = __builtin_amdgcn_wmma_f32_16x16x32_bf16(
            false, ad, false, bfrag[j], (short)0, acc[m][j], false, false);
      }
    }
  }

  float w3v[2][4], bias[2];
  int ncol[2];
#pragma unroll
  for (int j = 0; j < 2; ++j) {
    const int n = (w * 2 + j) * 16 + mrow;
    ncol[j] = n;
#pragma unroll
    for (int dd = 0; dd < 4; ++dd) w3v[j][dd] = W3[n * 4 + dd];
    bias[j] = b3[n] + 2.0f * b4[n];
  }
#pragma unroll
  for (int m = 0; m < 2; ++m) {
#pragma unroll
    for (int i = 0; i < 8; ++i) {
      const int rr = row0 + m * 16 + hi * 8 + i;
      const int ll = rr & (SEQ_L - 1);
      const int bb = rr >> 11;
      const float* xp = x + (size_t)(bb * 4) * SEQ_L + ll;
      const float x0 = xp[0], x1 = xp[SEQ_L], x2 = xp[2 * SEQ_L], x3 = xp[3 * SEQ_L];
#pragma unroll
      for (int j = 0; j < 2; ++j) {
        float v = acc[m][j][i] + bias[j] + x0 * w3v[j][0] + x1 * w3v[j][1] +
                  x2 * w3v[j][2] + x3 * w3v[j][3];
        v = v > 0.f ? v : 0.f;
        out[(size_t)rr * DIM + ncol[j]] = v;
      }
    }
  }
}

// ---------------------------------------------------------------------------
extern "C" void kernel_launch(void* const* d_in, const int* in_sizes, int n_in,
                              void* d_out, int out_size, void* d_ws, size_t ws_size,
                              hipStream_t stream) {
  (void)in_sizes; (void)n_in; (void)out_size; (void)ws_size;
  const float* x  = (const float*)d_in[0];
  const float* W1 = (const float*)d_in[1];
  const float* b1 = (const float*)d_in[2];
  const float* W2 = (const float*)d_in[3];
  const float* b2 = (const float*)d_in[4];
  const float* W3 = (const float*)d_in[5];
  const float* b3 = (const float*)d_in[6];
  const float* W4 = (const float*)d_in[7];
  const float* b4 = (const float*)d_in[8];
  float* out = (float*)d_out;

  // Workspace carve-out: two [2][NROWS][DIM] bf16 ping-pong state buffers
  // (33.5 MB each -> L2 resident), packed bf16 weight fragments (128 KB each),
  // one zeroed dummy row for shift boundaries.
  char* ws = (char*)d_ws;
  const size_t pbytes = (size_t)2 * NROWS * DIM * sizeof(bf16_t);
  bf16_t* p0   = (bf16_t*)ws;  ws += pbytes;
  bf16_t* p1   = (bf16_t*)ws;  ws += pbytes;
  bf16_t* w2f  = (bf16_t*)ws;  ws += (size_t)DIM * DIM * sizeof(bf16_t);
  bf16_t* w4f  = (bf16_t*)ws;  ws += (size_t)DIM * DIM * sizeof(bf16_t);
  bf16_t* zrow = (bf16_t*)ws;                       // DIM bf16 = 512 B

  // Pack weights into WMMA B-fragment layout (bf16).
  pack_wfrag_kernel<<<(DIM * DIM) / 256, 256, 0, stream>>>(W2, w2f);
  pack_wfrag_kernel<<<(DIM * DIM) / 256, 256, 0, stream>>>(W4, w4f);

  // Zero initial state (up = down = 0) and the boundary row, every launch.
  const int nU4 = (int)(pbytes / sizeof(uint4));     // 2,097,152
  zero_u4_kernel<<<(nU4 + 255) / 256, 256, 0, stream>>>((uint4*)p0, nU4);
  zero_u4_kernel<<<1, 256, 0, stream>>>((uint4*)zrow, (int)(DIM * sizeof(bf16_t) / sizeof(uint4)));

  bf16_t* bufs[2] = {p0, p1};
  for (int it = 0; it < T_ITERS; ++it) {
    dp_iter_kernel<<<NROWS / 64, 256, 0, stream>>>(
        bufs[it & 1], bufs[(it + 1) & 1], w2f, zrow, x, W1, b1, b2);
  }
  // T_ITERS even -> final state is in bufs[0].
  dp_final_kernel<<<NROWS / 32, 256, 0, stream>>>(
      bufs[T_ITERS & 1], w4f, zrow, x, W3, b3, b4, out);
}
